// Pointnet_SA_25967372272068
// MI455X (gfx1250) — matrix operand order repaired and checked by reading, not compiled, not run on hardware
//
#include <hip/hip_runtime.h>
#include <hip/hip_bf16.h>

// ---------------------------------------------------------------------------
// PointNet++ Set Abstraction for MI455X (gfx1250, wave32, WMMA)
//   kernel 1: farthest point sampling (1 block / batch, 32 waves, reg dists)
//   kernel 2: ball query (1 wave / query, ballot-based ordered radius scan)
//   kernel 3: gather + 3-layer MLP via v_wmma_f32_16x16x32_f16 + max-pool
// ---------------------------------------------------------------------------

#define B_       16
#define N_       4096
#define C_       64
#define NPOINT_  1024
#define NSAMPLE_ 32
#define RADIUS_  0.2f

typedef __attribute__((ext_vector_type(16))) _Float16 v16h;
typedef __attribute__((ext_vector_type(8)))  float    v8f;

// ======================= 1. Farthest Point Sampling ========================
__global__ __launch_bounds__(1024)
void fps_kernel(const float* __restrict__ xyz, float* __restrict__ out_xyz)
{
    __shared__ float s_val[32];
    __shared__ int   s_idx[32];
    __shared__ float s_cent[3];
    __shared__ int   s_best;

    const int b    = blockIdx.x;
    const int tid  = threadIdx.x;
    const int lane = tid & 31;
    const int wave = tid >> 5;
    const float* base = xyz + (size_t)b * N_ * 3;

    // 4 points per thread, strided for coalescing; dists live in registers.
    float px[4], py[4], pz[4], dist[4];
#pragma unroll
    for (int k = 0; k < 4; ++k) {
        int p = tid + k * 1024;
        px[k] = base[p * 3 + 0];
        py[k] = base[p * 3 + 1];
        pz[k] = base[p * 3 + 2];
        dist[k] = 1e10f;
    }
    if (tid == 0) {
        s_best    = 0;
        s_cent[0] = base[0]; s_cent[1] = base[1]; s_cent[2] = base[2];
    }
    __syncthreads();

    for (int it = 0; it < NPOINT_; ++it) {
        const float cx = s_cent[0], cy = s_cent[1], cz = s_cent[2];
        if (tid == 0) {
            float* o = out_xyz + ((size_t)b * NPOINT_ + it) * 3;
            o[0] = cx; o[1] = cy; o[2] = cz;
        }
        __syncthreads();   // everyone has read cent/best before overwrite

        float bv = -1.0f; int bi = 0;
#pragma unroll
        for (int k = 0; k < 4; ++k) {
            float dx = px[k] - cx, dy = py[k] - cy, dz = pz[k] - cz;
            float d  = dx * dx + dy * dy + dz * dz;
            dist[k]  = fminf(dist[k], d);
            int p = tid + k * 1024;
            if (dist[k] > bv || (dist[k] == bv && p < bi)) { bv = dist[k]; bi = p; }
        }
        // wave32 shuffle argmax (lowest index wins ties, matching jnp.argmax)
#pragma unroll
        for (int off = 16; off > 0; off >>= 1) {
            float ov = __shfl_xor(bv, off, 32);
            int   oi = __shfl_xor(bi, off, 32);
            if (ov > bv || (ov == bv && oi < bi)) { bv = ov; bi = oi; }
        }
        if (lane == 0) { s_val[wave] = bv; s_idx[wave] = bi; }
        __syncthreads();

        if (wave == 0) {
            bv = s_val[lane]; bi = s_idx[lane];
#pragma unroll
            for (int off = 16; off > 0; off >>= 1) {
                float ov = __shfl_xor(bv, off, 32);
                int   oi = __shfl_xor(bi, off, 32);
                if (ov > bv || (ov == bv && oi < bi)) { bv = ov; bi = oi; }
            }
            if (lane == 0) {
                s_best    = bi;
                s_cent[0] = base[bi * 3 + 0];
                s_cent[1] = base[bi * 3 + 1];
                s_cent[2] = base[bi * 3 + 2];
            }
        }
        __syncthreads();
    }
}

// ============================ 2. Ball Query ================================
// One wave32 per query. Sequential ballot scan reproduces the reference's
// "32 lowest in-radius indices, ascending, padded with the first" exactly.
__global__ __launch_bounds__(256)
void ball_query_kernel(const float* __restrict__ xyz,
                       const float* __restrict__ new_xyz,
                       int* __restrict__ ballidx)
{
    const int q    = blockIdx.x * 8 + (threadIdx.x >> 5);
    const int lane = threadIdx.x & 31;
    const int b    = q / NPOINT_;
    const float* base = xyz + (size_t)b * N_ * 3;
    const float qx = new_xyz[q * 3 + 0];
    const float qy = new_xyz[q * 3 + 1];
    const float qz = new_xyz[q * 3 + 2];
    const float r2 = RADIUS_ * RADIUS_;
    int* out = ballidx + (size_t)q * NSAMPLE_;

    int cnt = 0, first = 0;
    for (int n0 = 0; n0 < N_; n0 += 32) {
        const int p = n0 + lane;
        float dx = base[p * 3 + 0] - qx;
        float dy = base[p * 3 + 1] - qy;
        float dz = base[p * 3 + 2] - qz;
        bool pred = (dx * dx + dy * dy + dz * dz) <= r2;
        unsigned mask = (unsigned)__ballot(pred);     // wave32: low 32 bits
        if (cnt == 0 && mask) first = n0 + __ffs(mask) - 1;
        if (pred) {
            int pos = cnt + __popc(mask & ((1u << lane) - 1u));
            if (pos < NSAMPLE_) out[pos] = p;
        }
        cnt += __popc(mask);
        if (cnt >= NSAMPLE_) break;
    }
    if (cnt < NSAMPLE_ && lane >= cnt) out[lane] = first;  // pad with first
}

// ================= 3. Gather + MLP (WMMA) + Max-pool =======================
// 8 queries/block -> 256 rows -> 16 M-tiles; 8 waves, 2 M-tiles/wave.
#define QPB   8
#define ROWS  256
#define K0P   96     // 67 channels padded to 3 K-steps of 32

// Pack f32 weight (Kact x Nw) into per-fragment-contiguous f16 LDS layout so
// a lane's 16-half B fragment is 2x ds_load_b128.  B layout (32x16, 16-bit):
// lanes 0-15 hold K {0..7, 16..23} (N = lane), lanes 16-31: +8.
__device__ inline void pack_weights(_Float16* __restrict__ Wp,
                                    const float* __restrict__ W,
                                    int Kact, int Kpad, int Nw, int tid)
{
    const int NT = Nw >> 4;
    const int total = Kpad * Nw;
    for (int e = tid; e < total; e += 256) {
        int frag   = e >> 9;           // 512 halves per 32x16 fragment
        int within = e & 511;
        int lane   = within >> 4;
        int j      = within & 15;
        int kt = frag / NT, nt = frag % NT;
        int k  = kt * 32 + ((j >> 3) << 4) + ((lane >> 4) << 3) + (j & 7);
        int n  = nt * 16 + (lane & 15);
        float v = (k < Kact) ? W[k * Nw + n] : 0.0f;
        Wp[e] = (_Float16)v;
    }
}

// One layer: Xin (256 x Sin f16, row major) @ packed W -> relu(+bias) -> Xout.
// A layout (16x32, 16-bit): lanes 0-15: K {k0..k0+7, k0+16..k0+23};
// lanes 16-31: +8.  C/D: lanes 0-15 N=lane M=0..7, lanes 16-31 M=8..15.
__device__ inline void mlp_layer(const _Float16* __restrict__ Xin, int Sin,
                                 const _Float16* __restrict__ Wp,
                                 const float* __restrict__ bias,
                                 _Float16* __restrict__ Xout, int Sout,
                                 int KT, int NT, int wave, int lane)
{
    const int n  = lane & 15;
    const int mb = (lane >> 4) << 3;
    for (int mt = wave * 2; mt < wave * 2 + 2; ++mt) {
        const int m0 = mt * 16;
        const _Float16* arow = Xin + (m0 + (lane & 15)) * Sin + ((lane >> 4) << 3);
        for (int nt = 0; nt < NT; ++nt) {
            v8f acc = {};
            for (int kt = 0; kt < KT; ++kt) {
                const _Float16* ap = arow + kt * 32;
                v16h a, bf;
#pragma unroll
                for (int j = 0; j < 8; ++j) { a[j] = ap[j]; a[j + 8] = ap[16 + j]; }
                const _Float16* bp = Wp + (((kt * NT + nt) << 5) + lane) * 16;
#pragma unroll
                for (int j = 0; j < 16; ++j) bf[j] = bp[j];
                acc = __builtin_amdgcn_wmma_f32_16x16x32_f16(
                          false, a, false, bf, (short)0, acc, false, false);
            }
            const int   col = nt * 16 + n;
            const float bv  = bias[col];
#pragma unroll
            for (int i = 0; i < 8; ++i) {
                float v = acc[i] + bv;
                v = v > 0.0f ? v : 0.0f;                 // fused ReLU
                Xout[(m0 + mb + i) * Sout + col] = (_Float16)v;
            }
        }
    }
}

__global__ __launch_bounds__(256)
void group_mlp_max_kernel(const float* __restrict__ xyz,
                          const float* __restrict__ points,
                          const float* __restrict__ w0, const float* __restrict__ b0,
                          const float* __restrict__ w1, const float* __restrict__ b1,
                          const float* __restrict__ w2, const float* __restrict__ b2,
                          const int*   __restrict__ ballidx,
                          const float* __restrict__ new_xyz,
                          float* __restrict__ out_points)
{
    extern __shared__ char smem[];
    _Float16* X0  = (_Float16*)smem;                        // 256 x 96  (48 KB)
    _Float16* X1  = (_Float16*)(smem + 49152);              // 256 x 64  (32 KB)
    _Float16* H2  = (_Float16*)(smem + 49152 + 32768);      // 256 x 128 (64 KB)
    _Float16* W0p = (_Float16*)(smem + 49152 + 32768 + 65536);
    _Float16* W1p = W0p + 96 * 64;
    _Float16* W2p = W1p + 64 * 64;
    float*    bs0 = (float*)(W2p + 64 * 128);
    float*    bs1 = bs0 + 64;
    float*    bs2 = bs1 + 64;
    _Float16* X2  = X0;                                     // alias: X0 is dead

    const int tid  = threadIdx.x;
    const int lane = tid & 31;
    const int wave = tid >> 5;
    const int gq0  = blockIdx.x * QPB;
    const int b    = gq0 / NPOINT_;

    // ---- stage weights (packed) + biases into LDS ----
    pack_weights(W0p, w0, 67, K0P, 64,  tid);
    pack_weights(W1p, w1, 64, 64,  64,  tid);
    pack_weights(W2p, w2, 64, 64,  128, tid);
    if (tid < 64)  { bs0[tid] = b0[tid]; bs1[tid] = b1[tid]; }
    if (tid < 128) { bs2[tid] = b2[tid]; }

    // ---- gather grouped features: [xyz - center | points], f16, pad to 96 ----
    {
        const int r  = tid;               // one row per thread (256 == 256)
        const int q  = r >> 5;
        const int j  = r & 31;
        const int gq = gq0 + q;
        const int pidx = ballidx[(size_t)gq * NSAMPLE_ + j];
        const float* qc = new_xyz + (size_t)gq * 3;
        const float* pc = xyz + (size_t)b * N_ * 3 + (size_t)pidx * 3;
        _Float16* row = X0 + r * K0P;
        row[0] = (_Float16)(pc[0] - qc[0]);
        row[1] = (_Float16)(pc[1] - qc[1]);
        row[2] = (_Float16)(pc[2] - qc[2]);
        const float* f = points + (size_t)b * N_ * C_ + (size_t)pidx * C_;
#pragma unroll
        for (int c = 0; c < C_; ++c) row[3 + c] = (_Float16)f[c];
#pragma unroll
        for (int c = 67; c < K0P; ++c) row[c] = (_Float16)0.0f;
    }
    __syncthreads();

    // ---- 3 WMMA layers, ReLU fused on writeback ----
    mlp_layer(X0, K0P, W0p, bs0, X1, 64,  3, 4, wave, lane);   // 67->64
    __syncthreads();
    mlp_layer(X1, 64,  W1p, bs1, X2, 64,  2, 4, wave, lane);   // 64->64
    __syncthreads();
    mlp_layer(X2, 64,  W2p, bs2, H2, 128, 2, 8, wave, lane);   // 64->128
    __syncthreads();

    // ---- max over NSAMPLE (ReLU output >= 0, so 0-init is exact) ----
    for (int o = tid; o < QPB * 128; o += 256) {
        const int q  = o >> 7;
        const int ch = o & 127;
        const _Float16* colp = H2 + (size_t)(q * NSAMPLE_) * 128 + ch;
        float m = 0.0f;
#pragma unroll
        for (int j = 0; j < NSAMPLE_; ++j) m = fmaxf(m, (float)colp[j * 128]);
        out_points[(size_t)(gq0 + q) * 128 + ch] = m;
    }
}

// =============================== Launcher ==================================
extern "C" void kernel_launch(void* const* d_in, const int* in_sizes, int n_in,
                              void* d_out, int out_size, void* d_ws, size_t ws_size,
                              hipStream_t stream)
{
    const float* xyz    = (const float*)d_in[0];
    const float* points = (const float*)d_in[1];
    const float* w0     = (const float*)d_in[2];
    const float* b0     = (const float*)d_in[3];
    const float* w1     = (const float*)d_in[4];
    const float* b1     = (const float*)d_in[5];
    const float* w2     = (const float*)d_in[6];
    const float* b2     = (const float*)d_in[7];

    float* out_xyz    = (float*)d_out;                       // B*NPOINT*3
    float* out_points = out_xyz + (size_t)B_ * NPOINT_ * 3;  // B*NPOINT*128

    int* ballidx = (int*)d_ws;   // B*NPOINT*NSAMPLE ints = 2 MB scratch

    // 1) FPS: one 32-wave block per batch (sequential argmax chain)
    fps_kernel<<<B_, 1024, 0, stream>>>(xyz, out_xyz);

    // 2) Ball query: one wave per query
    ball_query_kernel<<<(B_ * NPOINT_) / 8, 256, 0, stream>>>(xyz, out_xyz, ballidx);

    // 3) Gather + WMMA MLP + max-pool: 8 queries per block, 181 KB LDS
    const size_t smem = 49152 + 32768 + 65536           // X0, X1, H2
                      + (96 * 64 + 64 * 64 + 64 * 128) * sizeof(_Float16)
                      + (64 + 64 + 128) * sizeof(float);
    group_mlp_max_kernel<<<(B_ * NPOINT_) / QPB, 256, smem, stream>>>(
        xyz, points, w0, b0, w1, b1, w2, b2, ballidx, out_xyz, out_points);
}
